// View2View_23098334118537
// MI455X (gfx1250) — compile-verified, matrix operand order checked by reading, not run on hardware
//
#include <hip/hip_runtime.h>
#include <math.h>

// ---------------- geometry constants ----------------
#define IMG   1024
#define NQ    300
#define NC    20
#define TOPK  100
#define EMBED 2700
#define KP    2720   // EMBED padded to multiple of 32 (85*32)
#define MP    224    // 2*TOPK=200 token rows padded (14 tiles of 16)
#define NKB   85     // KP/32 K-steps
#define MT    14     // MP/16 M-tiles
#define MBLK  7      // M-tiles per wave in GEMM1

typedef __attribute__((ext_vector_type(16))) __bf16 v16bf;
typedef __attribute__((ext_vector_type(8)))  float  v8f;

// Split a float pair into truncated-bf16 (hi) and residual-bf16 (lo) packed dwords.
__device__ __forceinline__ void split2(float x, float y, unsigned &hi, unsigned &lo) {
  unsigned ux = __float_as_uint(x), uy = __float_as_uint(y);
  unsigned hx = ux & 0xFFFF0000u,  hy = uy & 0xFFFF0000u;
  float rx = x - __uint_as_float(hx);
  float ry = y - __uint_as_float(hy);
  hi = (hx >> 16) | hy;
  lo = (__float_as_uint(rx) >> 16) | (__float_as_uint(ry) & 0xFFFF0000u);
}

__device__ __forceinline__ v8f wmma_bf16(v16bf a, v16bf b, v8f c) {
  return __builtin_amdgcn_wmma_f32_16x16x32_bf16(
      /*neg_a=*/false, a, /*neg_b=*/false, b,
      /*c_mod=*/(short)0, c, /*reuse_a=*/false, /*reuse_b=*/false);
}

union Frag { unsigned u[8]; uint4 q[2]; v16bf bf; };

// k-offset within a 16x16x32 bf16 fragment for packed pair p (0..7) and wave half (0/1).
__device__ __forceinline__ int frag_koff(int p, int half) {
  return (p < 4) ? (half * 8 + 2 * p) : (16 + half * 8 + 2 * (p - 4));
}

// ---------------------------------------------------------------------------
// K1: sigmoid -> top-100 over Q*C (descending, lowest-index tie-break),
// gather cxcywh -> xyxy, scale by IMG, round.  grid (side=2, b=2) x 512 thr.
// ---------------------------------------------------------------------------
__global__ void topk_boxes_kernel(const float* __restrict__ g_logits,
                                  const float* __restrict__ a_logits,
                                  const float* __restrict__ g_boxes,
                                  const float* __restrict__ a_boxes,
                                  float* __restrict__ boxes_out) {
  const int side = blockIdx.x;
  const int b    = blockIdx.y;
  const float* logits = (side == 0 ? g_logits : a_logits) + b * NQ * NC;
  const float* boxes  = (side == 0 ? g_boxes  : a_boxes ) + b * NQ * 4;
  __shared__ float sp[NQ * NC];
  __shared__ float rv[512];
  __shared__ int   ri[512];
  const int tid = threadIdx.x;
  for (int i = tid; i < NQ * NC; i += 512)
    sp[i] = 1.0f / (1.0f + expf(-logits[i]));
  __syncthreads();

  for (int t = 0; t < TOPK; ++t) {
    float bv = -__builtin_inff(); int bi = NQ * NC;
    for (int i = tid; i < NQ * NC; i += 512) {
      float c = sp[i];
      if (c > bv) { bv = c; bi = i; }
    }
    rv[tid] = bv; ri[tid] = bi;
    __syncthreads();
    for (int s = 256; s > 0; s >>= 1) {
      if (tid < s) {
        float ov = rv[tid + s]; int oi = ri[tid + s];
        if (ov > rv[tid] || (ov == rv[tid] && oi < ri[tid])) { rv[tid] = ov; ri[tid] = oi; }
      }
      __syncthreads();
    }
    if (tid == 0) {
      int idx = ri[0];
      sp[idx] = -__builtin_inff();
      int q = idx / NC;
      float cx = boxes[q * 4 + 0], cy = boxes[q * 4 + 1];
      float w  = boxes[q * 4 + 2], h  = boxes[q * 4 + 3];
      float* o = boxes_out + ((side * 2 + b) * TOPK + t) * 4;
      o[0] = rintf((cx - 0.5f * w) * (float)IMG);
      o[1] = rintf((cy - 0.5f * h) * (float)IMG);
      o[2] = rintf((cx + 0.5f * w) * (float)IMG);
      o[3] = rintf((cy + 0.5f * h) * (float)IMG);
    }
    __syncthreads();
  }
}

// ---------------------------------------------------------------------------
// K2: build padded token matrix tokens[side][m][k], m<200 & k<2700 real,
// rest zero.  Faithful permute: token (B_out,T) = patch(batch T&1,
// rank B_out*50 + T/2), e = c*900 + py*30 + px; plus pos_enc[T][e].
// ---------------------------------------------------------------------------
__global__ void build_tokens_kernel(const float* __restrict__ g_img,
                                    const float* __restrict__ a_img,
                                    const float* __restrict__ pos_enc,
                                    const float* __restrict__ boxes,
                                    float* __restrict__ tokens) {
  long long idx = (long long)blockIdx.x * blockDim.x + threadIdx.x;
  const long long total = 2LL * MP * KP;
  if (idx >= total) return;
  int k    = (int)(idx % KP);
  int m    = (int)((idx / KP) % MP);
  int side = (int)(idx / ((long long)KP * MP));
  float val = 0.0f;
  if (m < 2 * TOPK && k < EMBED) {
    int Bout = m / TOPK, T = m % TOPK;
    int bsrc = T & 1, trank = Bout * 50 + (T >> 1);
    int c = k / 900, p = k % 900, py = p / 30, px = p % 30;
    const float* bx = boxes + ((side * 2 + bsrc) * TOPK + trank) * 4;
    float x0 = bx[0], y0 = bx[1], x1 = bx[2], y1 = bx[3];
    float w = x1 - x0, h = y1 - y0;
    float patch = 0.0f;
    if (w > 0.0f && h > 0.0f) {
      const float* img = (side == 0 ? g_img : a_img) +
                         ((long long)bsrc * 3 + c) * IMG * IMG;
      float sy = fminf(fmaxf(((float)py + 0.5f) * h / 30.0f - 0.5f, 0.0f), fmaxf(h - 1.0f, 0.0f));
      float sx = fminf(fmaxf(((float)px + 0.5f) * w / 30.0f - 0.5f, 0.0f), fmaxf(w - 1.0f, 0.0f));
      float ay = y0 + sy, ax = x0 + sx;
      float yf = floorf(ay), xf = floorf(ax);
      float wy = ay - yf, wx = ax - xf;
      int y0i = (int)fminf(fmaxf(yf,        0.0f), (float)(IMG - 1));
      int y1i = (int)fminf(fmaxf(yf + 1.0f, 0.0f), (float)(IMG - 1));
      int x0i = (int)fminf(fmaxf(xf,        0.0f), (float)(IMG - 1));
      int x1i = (int)fminf(fmaxf(xf + 1.0f, 0.0f), (float)(IMG - 1));
      float v00 = img[y0i * IMG + x0i], v01 = img[y0i * IMG + x1i];
      float v10 = img[y1i * IMG + x0i], v11 = img[y1i * IMG + x1i];
      patch = (v00 * (1.0f - wx) + v01 * wx) * (1.0f - wy) +
              (v10 * (1.0f - wx) + v11 * wx) * wy;
    }
    val = patch + pos_enc[T * EMBED + k];
  }
  tokens[idx] = val;
}

// ---------------------------------------------------------------------------
// K2b: pre-split tokens into fragment-ordered bf16 hi/lo packs.
// One thread per (side, mtile, kbi, lane); writes lane's 8 dwords (2x uint4)
// for hi and lo.  Fragment layout matches v_wmma_f32_16x16x32_bf16 A-matrix.
// ---------------------------------------------------------------------------
__global__ void pack_tokens_kernel(const float* __restrict__ tokens,
                                   unsigned* __restrict__ packAhi,
                                   unsigned* __restrict__ packAlo) {
  int gid = blockIdx.x * blockDim.x + threadIdx.x;
  const int total = 2 * MT * NKB * 32;
  if (gid >= total) return;
  int lane  = gid & 31;
  int kbi   = (gid >> 5) % NKB;
  int mtile = (gid / (32 * NKB)) % MT;
  int side  = gid / (32 * NKB * MT);
  int half = lane >> 4, r = lane & 15;
  const float* A = tokens + ((long long)side * MP + mtile * 16 + r) * KP;
  Frag hi, lo;
#pragma unroll
  for (int p = 0; p < 8; ++p) {
    int k = kbi * 32 + frag_koff(p, half);
    float2 av = *(const float2*)(A + k);
    split2(av.x, av.y, hi.u[p], lo.u[p]);
  }
  long long base = ((((long long)side * MT + mtile) * NKB + kbi) * 32 + lane) * 8;
  *(uint4*)(packAhi + base)     = hi.q[0];
  *(uint4*)(packAhi + base + 4) = hi.q[1];
  *(uint4*)(packAlo + base)     = lo.q[0];
  *(uint4*)(packAlo + base + 4) = lo.q[1];
}

// ---------------------------------------------------------------------------
// K3: gtat[side] = tokens[side] @ W[side]^T.  One wave per 16-wide N column,
// register-blocked over MBLK=7 M-tiles: W is split to bf16 hi/lo once per
// K-step (branch-free: clamped address + AND mask) and reused by 21 WMMAs.
// grid (KP/16, MT/MBLK, 2), block 32.
// ---------------------------------------------------------------------------
__global__ void gemm_tokens_kernel(const unsigned* __restrict__ packAhi,
                                   const unsigned* __restrict__ packAlo,
                                   const float* __restrict__ Wg,
                                   const float* __restrict__ Wa,
                                   float* __restrict__ gtat) {
  const int lane = threadIdx.x & 31;
  const int half = lane >> 4, r = lane & 15;
  const int ntile = blockIdx.x, mgrp = blockIdx.y, side = blockIdx.z;
  const int n = ntile * 16 + r;                       // B column == W row
  const float* W = (side == 0 ? Wg : Wa);
  const float* Bsafe = W + (long long)min(n, EMBED - 1) * EMBED;
  const unsigned nmask = (n < EMBED) ? 0xFFFFFFFFu : 0u;

  v8f acc[MBLK];
#pragma unroll
  for (int mb = 0; mb < MBLK; ++mb) acc[mb] = (v8f){};

  const long long agrp0 =
      (((long long)side * MT + mgrp * MBLK) * NKB) * 32 * 8;  // first mtile, kbi=0

  for (int kbi = 0; kbi < NKB; ++kbi) {
    const int kb = kbi * 32;
    // ---- build B hi/lo fragments once (branch-free) ----
    Frag bh, bl;
#pragma unroll
    for (int p = 0; p < 8; ++p) {
      int k  = kb + frag_koff(p, half);
      int kc = min(k, EMBED - 2);                     // clamped, always valid
      float2 bv = *(const float2*)(Bsafe + kc);
      unsigned msk = (k < EMBED) ? nmask : 0u;
      split2(bv.x, bv.y, bh.u[p], bl.u[p]);
      bh.u[p] &= msk;
      bl.u[p] &= msk;
    }
    __builtin_prefetch(Bsafe + kb + 32, 0, 3);        // next K-panel of W

    // ---- 7 M-tiles reuse the B fragments ----
#pragma unroll
    for (int mb = 0; mb < MBLK; ++mb) {
      long long abase = agrp0 + (((long long)mb * NKB + kbi) * 32 + lane) * 8;
      Frag ah, al;
      ah.q[0] = *(const uint4*)(packAhi + abase);
      ah.q[1] = *(const uint4*)(packAhi + abase + 4);
      al.q[0] = *(const uint4*)(packAlo + abase);
      al.q[1] = *(const uint4*)(packAlo + abase + 4);
      acc[mb] = wmma_bf16(ah.bf, bh.bf, acc[mb]);
      acc[mb] = wmma_bf16(ah.bf, bl.bf, acc[mb]);
      acc[mb] = wmma_bf16(al.bf, bh.bf, acc[mb]);
    }
  }

  float* D = gtat + (long long)side * MP * KP;
#pragma unroll
  for (int mb = 0; mb < MBLK; ++mb) {
    int mtile = mgrp * MBLK + mb;
#pragma unroll
    for (int v = 0; v < 8; ++v) {
      int row = mtile * 16 + v + 8 * half;
      int col = ntile * 16 + r;
      D[(long long)row * KP + col] = acc[mb][v];
    }
  }
}

// ---------------------------------------------------------------------------
// K4: sim[b][i][j] = (gt[b,i] . at[b,j]) / sqrt(384), split-bf16 WMMA.
// grid (7, 7, 2), block 32.  Padded K columns/rows of gtat are zero; loop is
// branch-free.
// ---------------------------------------------------------------------------
__global__ void sim_gemm_kernel(const float* __restrict__ gtat,
                                float* __restrict__ sim) {
  const int lane = threadIdx.x & 31;
  const int half = lane >> 4, r = lane & 15;
  const int ntile = blockIdx.x, mtile = blockIdx.y, b = blockIdx.z;
  const float* gt = gtat;
  const float* at = gtat + (long long)MP * KP;
  const float* A  = gt + ((long long)b * TOPK + mtile * 16 + r) * KP;
  const float* Bp = at + ((long long)b * TOPK + ntile * 16 + r) * KP;
  v8f acc = {};
  for (int kb = 0; kb < KP; kb += 32) {
    Frag ah, al, bh, bl;
#pragma unroll
    for (int p = 0; p < 8; ++p) {
      int k = kb + frag_koff(p, half);
      float2 av = *(const float2*)(A  + k);
      float2 bv = *(const float2*)(Bp + k);
      split2(av.x, av.y, ah.u[p], al.u[p]);
      split2(bv.x, bv.y, bh.u[p], bl.u[p]);
    }
    acc = wmma_bf16(ah.bf, bh.bf, acc);
    acc = wmma_bf16(ah.bf, bl.bf, acc);
    acc = wmma_bf16(al.bf, bh.bf, acc);
  }
  const float scale = 0.05103103630798288f; // 1/sqrt(384)
#pragma unroll
  for (int v = 0; v < 8; ++v) {
    int i = mtile * 16 + v + 8 * half;
    int j = ntile * 16 + r;
    if (i < TOPK && j < TOPK)
      sim[((long long)b * TOPK + i) * TOPK + j] = acc[v] * scale;
  }
}

// ---------------------------------------------------------------------------
// K5: softmax over axis=1 (the i/query dim).  One block per (b, j).
// ---------------------------------------------------------------------------
__global__ void softmax_kernel(const float* __restrict__ sim,
                               float* __restrict__ out) {
  const int b = blockIdx.x / TOPK, j = blockIdx.x % TOPK;
  const int tid = threadIdx.x;
  __shared__ float red[128];
  float v = (tid < TOPK) ? sim[((long long)b * TOPK + tid) * TOPK + j] : -__builtin_inff();
  red[tid] = v;
  __syncthreads();
  for (int s = 64; s > 0; s >>= 1) {
    if (tid < s) red[tid] = fmaxf(red[tid], red[tid + s]);
    __syncthreads();
  }
  float mx = red[0];
  __syncthreads();
  float e = (tid < TOPK) ? expf(v - mx) : 0.0f;
  red[tid] = e;
  __syncthreads();
  for (int s = 64; s > 0; s >>= 1) {
    if (tid < s) red[tid] += red[tid + s];
    __syncthreads();
  }
  float sum = red[0];
  if (tid < TOPK) out[((long long)b * TOPK + tid) * TOPK + j] = e / sum;
}

// ---------------------------------------------------------------------------
extern "C" void kernel_launch(void* const* d_in, const int* in_sizes, int n_in,
                              void* d_out, int out_size, void* d_ws, size_t ws_size,
                              hipStream_t stream) {
  const float* g_samples = (const float*)d_in[0];
  const float* a_samples = (const float*)d_in[1];
  const float* g_logits  = (const float*)d_in[2];
  const float* a_logits  = (const float*)d_in[3];
  const float* g_boxes   = (const float*)d_in[4];
  const float* a_boxes   = (const float*)d_in[5];
  const float* W_ground  = (const float*)d_in[6];
  const float* W_aerial  = (const float*)d_in[7];
  const float* pos_enc   = (const float*)d_in[8];

  // workspace layout (dwords, all segment offsets 16B-aligned):
  // boxes | tokens f32 | gtat | sim | packAhi | packAlo
  float* ws     = (float*)d_ws;
  float* boxes  = ws;                               // 1600 used, reserve 2048
  float* tokens = ws + 2048;                        // 2*224*2720 = 1218560
  float* gtat   = tokens + 2LL * MP * KP;           // 1218560
  float* sim    = gtat   + 2LL * MP * KP;           // 20000 used, reserve 20480
  unsigned* packAhi = (unsigned*)(sim + 20480);     // 2*14*85*32*8 = 609280
  unsigned* packAlo = packAhi + 2LL * MT * NKB * 32 * 8;
  (void)in_sizes; (void)n_in; (void)out_size; (void)ws_size;

  topk_boxes_kernel<<<dim3(2, 2), 512, 0, stream>>>(g_logits, a_logits,
                                                    g_boxes, a_boxes, boxes);

  const long long tok_total = 2LL * MP * KP;
  build_tokens_kernel<<<(int)((tok_total + 255) / 256), 256, 0, stream>>>(
      g_samples, a_samples, pos_enc, boxes, tokens);

  const int pack_total = 2 * MT * NKB * 32;
  pack_tokens_kernel<<<(pack_total + 255) / 256, 256, 0, stream>>>(
      tokens, packAhi, packAlo);

  gemm_tokens_kernel<<<dim3(KP / 16, MT / MBLK, 2), 32, 0, stream>>>(
      packAhi, packAlo, W_ground, W_aerial, gtat);

  sim_gemm_kernel<<<dim3(7, 7, 2), 32, 0, stream>>>(gtat, sim);

  softmax_kernel<<<2 * TOPK, 128, 0, stream>>>(sim, (float*)d_out);
}
